// Block_90469191123033
// MI455X (gfx1250) — matrix operand order repaired and checked by reading, not compiled, not run on hardware
//
#include <hip/hip_runtime.h>

typedef __bf16 bf16_t;
typedef __attribute__((ext_vector_type(16))) __bf16 v16bf;
typedef __attribute__((ext_vector_type(8)))  __bf16 v8bf;
typedef __attribute__((ext_vector_type(8)))  float  v8f;

#define NE_   1024
#define NH_   16
#define TB_   2
#define TT_   2048
#define ROWS_ (TB_*TT_)   // 4096
#define EPSF  1.1920929e-07f
#define PARTMAX 16384

// ---------------------------------------------------------------------------
// Tensor Data Mover (CDNA5): this toolchain exposes the 6-arg builtin form
// (uint32x4 g0, int32x8 g1, int32x4 g2, int32x4 g3, int32x8 g4, i32 cpol).
// Falls back to cooperative vector loads when the builtins are unavailable.
// ---------------------------------------------------------------------------
#if defined(__gfx1250__) && __has_builtin(__builtin_amdgcn_tensor_load_to_lds) && \
    __has_builtin(__builtin_amdgcn_s_wait_tensorcnt)
#define USE_TDM 1
#else
#define USE_TDM 0
#endif

#if USE_TDM
typedef __attribute__((ext_vector_type(4))) unsigned int v4u;
typedef __attribute__((ext_vector_type(8))) int v8i_;
typedef __attribute__((ext_vector_type(4))) int v4i_;

// 2D tile DMA: global (row-major, row_stride elems of bf16) -> LDS.
// Optional LDS padding: pad_amount+1 DWORDs inserted every 2^(pad_interval+1) DWORDs.
__device__ static inline void tdm_load_2d(const bf16_t* gsrc, unsigned lds_off_bytes,
                                          unsigned tile_k, unsigned tile_rows,
                                          unsigned row_stride, unsigned tensor_d0,
                                          unsigned tensor_d1, int pad_enable,
                                          int pad_interval, int pad_amount) {
  unsigned long long ga = (unsigned long long)(const void*)gsrc;
  v4u g0;
  g0[0] = 1u;                                    // count=1, user descriptor
  g0[1] = lds_off_bytes;                         // lds_addr
  g0[2] = (unsigned)(ga & 0xffffffffu);          // global_addr[31:0]
  g0[3] = (unsigned)((ga >> 32) & 0x1ffffffu) | 0x80000000u;  // addr[56:32] | type=2
  v8i_ g1;
  g1[0] = (int)((1u << 16) |                     // data_size = 2 bytes
                (pad_enable ? (1u << 20) : 0u) |
                (((unsigned)pad_interval & 7u) << 22) |
                (((unsigned)pad_amount & 127u) << 25));
  g1[1] = (int)((tensor_d0 & 0xffffu) << 16);    // tensor_dim0[15:0] @ bits 63:48
  g1[2] = (int)((tensor_d0 >> 16) | ((tensor_d1 & 0xffffu) << 16));
  g1[3] = (int)((tensor_d1 >> 16) | ((tile_k & 0xffffu) << 16));  // tile_dim0
  g1[4] = (int)(tile_rows & 0xffffu);            // tile_dim1 (tile_dim2 = 0)
  g1[5] = (int)row_stride;                       // tensor_dim0_stride[31:0]
  g1[6] = 0;                                     // stride[47:32] | dim1_stride lo
  g1[7] = 0;
  v4i_ z4 = {0, 0, 0, 0};
  v8i_ z8 = {0, 0, 0, 0, 0, 0, 0, 0};
  __builtin_amdgcn_tensor_load_to_lds(g0, g1, z4, z4, z8, 0);
}
#endif

// ---------------------------------------------------------------------------
// GEMM: C[M,N] = scale * A[M,K] * B[N,K]^T (+ add), A/B bf16 row-major K-contig
// 128x128 block tile, 8 waves, each wave 32(M)x64(N) = 2x4 WMMA 16x16 tiles.
// TDM path: double-buffered LDS tiles, DMA issued by wave 0, TENSORcnt sync.
// ---------------------------------------------------------------------------
__global__ __launch_bounds__(256)
void gemm_wmma(float* __restrict__ C, const bf16_t* __restrict__ A,
               const bf16_t* __restrict__ Bm, int M, int N, int K,
               float scale, const float* __restrict__ add) {
  __shared__ __attribute__((aligned(32))) bf16_t la[2][128 * 48];
  __shared__ __attribute__((aligned(32))) bf16_t lb[2][128 * 48];
  const int tid  = threadIdx.x;
  const int wid  = tid >> 5;
  const int lane = tid & 31;
  const int lr   = lane & 15, lh = lane >> 4;
  const int m0 = blockIdx.y * 128;
  const int n0 = blockIdx.x * 128;
  const int wm = (wid & 3) * 32;   // 4 waves along M
  const int wn = (wid >> 2) * 64;  // 2 waves along N
  const int ldr = tid >> 1;        // 0..127
  const int ldc = (tid & 1) * 16;  // 0 / 16

  v8f acc[2][4];
  for (int i = 0; i < 2; i++)
    for (int j = 0; j < 4; j++)
      for (int e = 0; e < 8; e++) acc[i][j][e] = 0.f;

  for (int k0 = 0; k0 < K; k0 += 32) {
    const int cur = (k0 >> 5) & 1;
#if USE_TDM
    if (k0 == 0 && wid == 0) {
      // prologue: DMA first tiles. LDS rows padded 32->48 elems:
      // 16 DWORDs data then 8 DWORDs pad => interval code 3, amount code 7.
      tdm_load_2d(A + (size_t)m0 * K, (unsigned)(size_t)(void*)la[0],
                  32, 128, (unsigned)K, (unsigned)K, (unsigned)M, 1, 3, 7);
      tdm_load_2d(Bm + (size_t)n0 * K, (unsigned)(size_t)(void*)lb[0],
                  32, 128, (unsigned)K, (unsigned)K, (unsigned)N, 1, 3, 7);
    }
    if (wid == 0) __builtin_amdgcn_s_wait_tensorcnt(0);
    __syncthreads();
    if (k0 + 32 < K && wid == 0) {
      const int nxt = cur ^ 1;
      tdm_load_2d(A + (size_t)m0 * K + (k0 + 32), (unsigned)(size_t)(void*)la[nxt],
                  32, 128, (unsigned)K, (unsigned)K, (unsigned)M, 1, 3, 7);
      tdm_load_2d(Bm + (size_t)n0 * K + (k0 + 32), (unsigned)(size_t)(void*)lb[nxt],
                  32, 128, (unsigned)K, (unsigned)K, (unsigned)N, 1, 3, 7);
    }
#else
    __syncthreads();
    *(v16bf*)(la[cur] + ldr * 48 + ldc) =
        *(const v16bf*)(A + (size_t)(m0 + ldr) * K + k0 + ldc);
    *(v16bf*)(lb[cur] + ldr * 48 + ldc) =
        *(const v16bf*)(Bm + (size_t)(n0 + ldr) * K + k0 + ldc);
    if (k0 + 32 < K) {  // emits global_prefetch_b8 on gfx1250
      __builtin_prefetch(A + (size_t)(m0 + ldr) * K + k0 + 32 + ldc, 0, 1);
      __builtin_prefetch(Bm + (size_t)(n0 + ldr) * K + k0 + 32 + ldc, 0, 1);
    }
    __syncthreads();
#endif

    v16bf af[2], bfr[4];
#pragma unroll
    for (int sm = 0; sm < 2; sm++) {
      const bf16_t* p = la[cur] + (wm + sm * 16 + lr) * 48 + lh * 16;
      v8bf lo = *(const v8bf*)p;
      v8bf hi = *(const v8bf*)(p + 8);
      v16bf f;
#pragma unroll
      for (int i = 0; i < 8; i++) { f[i] = lo[i]; f[8 + i] = hi[i]; }
      af[sm] = f;
    }
#pragma unroll
    for (int sn = 0; sn < 4; sn++)
      bfr[sn] = *(const v16bf*)(lb[cur] + (wn + sn * 16 + lr) * 48 + lh * 16);

#pragma unroll
    for (int sm = 0; sm < 2; sm++)
#pragma unroll
      for (int sn = 0; sn < 4; sn++)
        acc[sm][sn] = __builtin_amdgcn_wmma_f32_16x16x32_bf16(
            false, af[sm], false, bfr[sn], (short)0, acc[sm][sn], false, false);
  }

  for (int sm = 0; sm < 2; sm++)
    for (int sn = 0; sn < 4; sn++) {
      int mb = m0 + wm + sm * 16;
      int nb = n0 + wn + sn * 16;
#pragma unroll
      for (int r = 0; r < 8; r++) {
        int m = mb + r + lh * 8;
        int n = nb + lr;
        size_t idx = (size_t)m * N + n;
        float v = acc[sm][sn][r] * scale;
        if (add) v += add[idx];
        C[idx] = v;
      }
    }
}

// ---------------------------------------------------------------------------
// Flash attention per (b,h): 128 q-rows/block (16 per wave), causal, bf16 WMMA.
// ---------------------------------------------------------------------------
__global__ __launch_bounds__(256)
void flash_attn(float* __restrict__ Y, const bf16_t* __restrict__ Qh,
                const bf16_t* __restrict__ Kh, const bf16_t* __restrict__ Vh) {
  __shared__ __attribute__((aligned(32))) bf16_t kt[64 * 64];
  __shared__ __attribute__((aligned(32))) bf16_t vt[64 * 64];
  __shared__ __attribute__((aligned(32))) bf16_t pbuf[8][16 * 64];
  const int tid = threadIdx.x, wid = tid >> 5, lane = tid & 31;
  const int lr = lane & 15, lh = lane >> 4;
  const int b = blockIdx.z, h = blockIdx.y, qbase = blockIdx.x * 128;
  const int q0 = qbase + wid * 16;
  const size_t headoff = ((size_t)b * NH_ + h) * TT_;

  v16bf qa[2];
  {
    const bf16_t* qp = Qh + (headoff + q0 + lr) * 64;
#pragma unroll
    for (int f = 0; f < 2; f++) {
      v8bf lo = *(const v8bf*)(qp + f * 32 + lh * 16);
      v8bf hi = *(const v8bf*)(qp + f * 32 + lh * 16 + 8);
      v16bf fr;
#pragma unroll
      for (int i = 0; i < 8; i++) { fr[i] = lo[i]; fr[8 + i] = hi[i]; }
      qa[f] = fr;
    }
  }

  float m_i[8], l_i[8];
  v8f o[4];
  for (int r = 0; r < 8; r++) { m_i[r] = -3.0e38f; l_i[r] = 0.f; }
  for (int nt = 0; nt < 4; nt++)
    for (int e = 0; e < 8; e++) o[nt][e] = 0.f;

  const int ldr = tid >> 2, ldc = (tid & 3) * 16;
  for (int j0 = 0; j0 < qbase + 128; j0 += 64) {
    __syncthreads();
#if USE_TDM
    if (wid == 0) {
      tdm_load_2d(Kh + (headoff + j0) * 64, (unsigned)(size_t)(void*)kt,
                  64, 64, 64, 64, (unsigned)TT_, 0, 0, 0);
      tdm_load_2d(Vh + (headoff + j0) * 64, (unsigned)(size_t)(void*)vt,
                  64, 64, 64, 64, (unsigned)TT_, 0, 0, 0);
      __builtin_amdgcn_s_wait_tensorcnt(0);
    }
    __syncthreads();
#else
    *(v16bf*)(kt + ldr * 64 + ldc) =
        *(const v16bf*)(Kh + (headoff + j0 + ldr) * 64 + ldc);
    *(v16bf*)(vt + ldr * 64 + ldc) =
        *(const v16bf*)(Vh + (headoff + j0 + ldr) * 64 + ldc);
    __syncthreads();
#endif

    float sv[4][8];
#pragma unroll
    for (int jj = 0; jj < 4; jj++) {
      v8f s;
      for (int e = 0; e < 8; e++) s[e] = 0.f;
#pragma unroll
      for (int f = 0; f < 2; f++) {
        v16bf bk = *(const v16bf*)(kt + (jj * 16 + lr) * 64 + f * 32 + lh * 16);
        s = __builtin_amdgcn_wmma_f32_16x16x32_bf16(false, qa[f], false, bk,
                                                    (short)0, s, false, false);
      }
#pragma unroll
      for (int r = 0; r < 8; r++) {
        int qi = q0 + r + lh * 8;
        int ki = j0 + jj * 16 + lr;
        sv[jj][r] = (ki <= qi) ? s[r] * 0.125f : -3.0e38f;
      }
    }
    // online softmax over this 64-key block (rows live across 16-lane halves)
#pragma unroll
    for (int r = 0; r < 8; r++) {
      float rmax = sv[0][r];
      for (int jj = 1; jj < 4; jj++) rmax = fmaxf(rmax, sv[jj][r]);
      for (int off = 1; off < 16; off <<= 1)
        rmax = fmaxf(rmax, __shfl_xor(rmax, off, 16));
      float mnew = fmaxf(m_i[r], rmax);
      float corr = __expf(m_i[r] - mnew);
      float rsum = 0.f;
      for (int jj = 0; jj < 4; jj++) {
        float p = __expf(sv[jj][r] - mnew);
        sv[jj][r] = p;
        rsum += p;
      }
      for (int off = 1; off < 16; off <<= 1) rsum += __shfl_xor(rsum, off, 16);
      l_i[r] = l_i[r] * corr + rsum;
      m_i[r] = mnew;
      for (int nt = 0; nt < 4; nt++) o[nt][r] *= corr;
    }
#pragma unroll
    for (int jj = 0; jj < 4; jj++)
#pragma unroll
      for (int r = 0; r < 8; r++)
        pbuf[wid][(r + lh * 8) * 64 + jj * 16 + lr] = (bf16_t)sv[jj][r];
    __syncthreads();
    // P * V
#pragma unroll
    for (int kc = 0; kc < 2; kc++) {
      v16bf pa;
      {
        const bf16_t* pp = pbuf[wid] + lr * 64 + kc * 32 + lh * 16;
        v8bf lo = *(const v8bf*)pp, hi = *(const v8bf*)(pp + 8);
#pragma unroll
        for (int i = 0; i < 8; i++) { pa[i] = lo[i]; pa[8 + i] = hi[i]; }
      }
#pragma unroll
      for (int nt = 0; nt < 4; nt++) {
        v16bf bv;
        int d = nt * 16 + lr;
#pragma unroll
        for (int i = 0; i < 16; i++) bv[i] = vt[(kc * 32 + lh * 16 + i) * 64 + d];
        o[nt] = __builtin_amdgcn_wmma_f32_16x16x32_bf16(false, pa, false, bv,
                                                        (short)0, o[nt], false, false);
      }
    }
  }
#pragma unroll
  for (int r = 0; r < 8; r++) {
    int t = q0 + r + lh * 8;
    float inv = 1.f / l_i[r];
    size_t yb = ((size_t)b * TT_ + t) * 1024 + h * 64;
    for (int nt = 0; nt < 4; nt++) Y[yb + nt * 16 + lr] = o[nt][r] * inv;
  }
}

// ---------------------------------------------------------------------------
// Elementwise / row kernels
// ---------------------------------------------------------------------------
__global__ __launch_bounds__(256)
void rmsnorm_rows_bf16(const float* __restrict__ X, bf16_t* __restrict__ O, int D) {
  __shared__ float sred[8];
  size_t row = blockIdx.x;
  const float* x = X + row * (size_t)D;
  bf16_t* o = O + row * (size_t)D;
  float ss = 0.f;
  for (int i = threadIdx.x; i < D; i += 256) { float v = x[i]; ss += v * v; }
  for (int off = 1; off < 32; off <<= 1) ss += __shfl_xor(ss, off, 32);
  if ((threadIdx.x & 31) == 0) sred[threadIdx.x >> 5] = ss;
  __syncthreads();
  float tot = 0.f;
  for (int w = 0; w < 8; w++) tot += sred[w];
  float r = rsqrtf(tot / D + EPSF);
  for (int i = threadIdx.x; i < D; i += 256) o[i] = (bf16_t)(x[i] * r);
}

__global__ __launch_bounds__(256)
void rmsnorm_rows_f32(float* __restrict__ X, int D) {
  __shared__ float sred[8];
  size_t row = blockIdx.x;
  float* x = X + row * (size_t)D;
  float ss = 0.f;
  for (int i = threadIdx.x; i < D; i += 256) { float v = x[i]; ss += v * v; }
  for (int off = 1; off < 32; off <<= 1) ss += __shfl_xor(ss, off, 32);
  if ((threadIdx.x & 31) == 0) sred[threadIdx.x >> 5] = ss;
  __syncthreads();
  float tot = 0.f;
  for (int w = 0; w < 8; w++) tot += sred[w];
  float r = rsqrtf(tot / D + EPSF);
  for (int i = threadIdx.x; i < D; i += 256) x[i] *= r;
}

__global__ void pack_bf16(bf16_t* __restrict__ dst, const float* __restrict__ src,
                          long total, int K, long rs, long cs) {
  long i = blockIdx.x * (long)blockDim.x + threadIdx.x;
  if (i >= total) return;
  long m = i / K, k = i % K;
  dst[i] = (bf16_t)src[m * rs + k * cs];
}

__global__ __launch_bounds__(256)
void aft_norm(float* __restrict__ qkv, float* __restrict__ wbuf,
              float* __restrict__ wvbuf) {
  __shared__ float sred[16];
  size_t row = blockIdx.x;
  float* base = qkv + row * 3072;
  float sq = 0.f, sk = 0.f;
  for (int i = threadIdx.x; i < 1024; i += 256) {
    float q = base[i];        sq += q * q;
    float k = base[1024 + i]; sk += k * k;
  }
  for (int off = 1; off < 32; off <<= 1) {
    sq += __shfl_xor(sq, off, 32);
    sk += __shfl_xor(sk, off, 32);
  }
  if ((threadIdx.x & 31) == 0) {
    sred[threadIdx.x >> 5] = sq;
    sred[8 + (threadIdx.x >> 5)] = sk;
  }
  __syncthreads();
  float tq = 0.f, tk = 0.f;
  for (int w = 0; w < 8; w++) { tq += sred[w]; tk += sred[8 + w]; }
  float rq = rsqrtf(tq / 1024.f + EPSF), rk = rsqrtf(tk / 1024.f + EPSF);
  for (int i = threadIdx.x; i < 1024; i += 256) {
    float qn = base[i] * rq;
    base[i] = qn;  // keep rmsnorm'd q in place for the scan kernel
    float w = __expf(base[1024 + i] * rk);
    wbuf[row * 1024 + i] = w;
    wvbuf[row * 1024 + i] = w * base[2048 + i];
  }
}

__global__ void aft_scan(const float* __restrict__ qkv, const float* __restrict__ wbuf,
                         const float* __restrict__ wvbuf, float* __restrict__ y) {
  int j = blockIdx.x * blockDim.x + threadIdx.x;  // (b, channel)
  int b = j >> 10, c = j & 1023;
  float wc = 0.f, kvc = 0.f;
  for (int t = 0; t < TT_; t++) {
    size_t r = (size_t)(b * TT_ + t);
    wc += wbuf[r * 1024 + c];
    kvc += wvbuf[r * 1024 + c];
    float qn = qkv[r * 3072 + c];
    float sig = 1.f / (1.f + __expf(-qn));
    y[r * 1024 + c] = sig * kvc / (wc + 1e-6f);
  }
}

__global__ void tea_prep(const float* __restrict__ qkv, const float* __restrict__ cosT,
                         const float* __restrict__ sinT, bf16_t* __restrict__ Qh,
                         bf16_t* __restrict__ Kh, bf16_t* __restrict__ Vh) {
  int gid = blockIdx.x * blockDim.x + threadIdx.x;
  if (gid >= ROWS_ * NH_) return;
  int h = gid & 15;
  int bt = gid >> 4;
  int t = bt % TT_, b = bt / TT_;
  const float* src = qkv + (size_t)bt * 3072 + h * 192;
  const float* cs = cosT + t * 32;
  const float* sn = sinT + t * 32;
  float q[64], k[64];
  for (int d = 0; d < 32; d++) {
    float c = cs[d], s = sn[d];
    float q1 = src[d],      q2 = src[32 + d];
    q[d] = q1 * c + q2 * s; q[32 + d] = -q1 * s + q2 * c;
    float k1 = src[64 + d], k2 = src[96 + d];
    k[d] = k1 * c + k2 * s; k[32 + d] = -k1 * s + k2 * c;
  }
  float sq = 0.f, sk = 0.f;
  for (int d = 0; d < 64; d++) { sq += q[d] * q[d]; sk += k[d] * k[d]; }
  float rq = rsqrtf(sq / 64.f + EPSF), rk = rsqrtf(sk / 64.f + EPSF);
  size_t o = (((size_t)b * NH_ + h) * TT_ + t) * 64;
  for (int d = 0; d < 64; d++) {
    Qh[o + d] = (bf16_t)(q[d] * rq);
    Kh[o + d] = (bf16_t)(k[d] * rk);
    Vh[o + d] = (bf16_t)src[128 + d];
  }
}

__global__ void u_silu_pack(const float* __restrict__ uv, bf16_t* __restrict__ sbf) {
  long i = blockIdx.x * (long)blockDim.x + threadIdx.x;
  if (i >= (long)ROWS_ * 1024) return;
  long m = i >> 10, c = i & 1023;
  float u = uv[m * 2048 + c], v = uv[m * 2048 + 1024 + c];
  float s = v / (1.f + __expf(-v));
  sbf[i] = (bf16_t)(u * s);
}

__global__ void silu_pack(const float* __restrict__ src, bf16_t* __restrict__ dst,
                          long n) {
  long i = blockIdx.x * (long)blockDim.x + threadIdx.x;
  if (i >= n) return;
  float x = src[i];
  dst[i] = (bf16_t)(x / (1.f + __expf(-x)));
}

// w += n (with optional ws index remap); deterministic per-block partial sums
__global__ __launch_bounds__(256)
void add_reduce(float* __restrict__ w, const float* __restrict__ nsrc, long cnt,
                int remap, float* __restrict__ part) {
  __shared__ float s0[8], s1[8];
  long i = blockIdx.x * (long)blockDim.x + threadIdx.x;
  float t = 0.f, t2 = 0.f;
  if (i < cnt) {
    long ni = remap ? ((i & 2047) * 1024 + (i >> 11)) : i;
    float val = w[i] + nsrc[ni];
    w[i] = val;
    t = val;
    t2 = val * val;
  }
  for (int off = 1; off < 32; off <<= 1) {
    t += __shfl_xor(t, off, 32);
    t2 += __shfl_xor(t2, off, 32);
  }
  if ((threadIdx.x & 31) == 0) {
    s0[threadIdx.x >> 5] = t;
    s1[threadIdx.x >> 5] = t2;
  }
  __syncthreads();
  if (threadIdx.x == 0) {
    float a = 0.f, b = 0.f;
    for (int k2 = 0; k2 < 8; k2++) { a += s0[k2]; b += s1[k2]; }
    part[blockIdx.x] = a;
    part[PARTMAX + blockIdx.x] = b;
  }
}

__global__ __launch_bounds__(256)
void final_reduce(const float* __restrict__ part, int nblocks, float* __restrict__ red) {
  __shared__ float s0[8], s1[8];
  float a = 0.f, b = 0.f;
  for (int i = threadIdx.x; i < nblocks; i += 256) {
    a += part[i];
    b += part[PARTMAX + i];
  }
  for (int off = 1; off < 32; off <<= 1) {
    a += __shfl_xor(a, off, 32);
    b += __shfl_xor(b, off, 32);
  }
  if ((threadIdx.x & 31) == 0) { s0[threadIdx.x >> 5] = a; s1[threadIdx.x >> 5] = b; }
  __syncthreads();
  if (threadIdx.x == 0) {
    float ta = 0.f, tb = 0.f;
    for (int w = 0; w < 8; w++) { ta += s0[w]; tb += s1[w]; }
    red[0] = ta;
    red[1] = tb;
  }
}

__global__ void scale_w(float* __restrict__ w, long cnt, const float* __restrict__ red,
                        const float* __restrict__ tao, int idx, float invsqrtn) {
  long i = blockIdx.x * (long)blockDim.x + threadIdx.x;
  if (i >= cnt) return;
  float sum = red[0], sumsq = red[1];
  float var = (sumsq - sum * sum / (float)cnt) / (float)(cnt - 1);
  float sd = sqrtf(fmaxf(var, 0.f));
  float a = fabsf(tao[idx]);
  a = fminf(fmaxf(a, 1e-8f), 1.f);
  w[i] *= invsqrtn / (sd + 1e-8f) * a;
}

// ---------------------------------------------------------------------------
// Host orchestration
// ---------------------------------------------------------------------------
static void run_gemm(float* C, const bf16_t* A, const bf16_t* B, int M, int N, int K,
                     float scale, const float* add, hipStream_t s) {
  dim3 g(N / 128, M / 128);
  gemm_wmma<<<g, 256, 0, s>>>(C, A, B, M, N, K, scale, add);
}
static void run_pack(bf16_t* dst, const float* src, long total, int K, long rs, long cs,
                     hipStream_t s) {
  long blocks = (total + 255) / 256;
  pack_bf16<<<dim3((unsigned)blocks), 256, 0, s>>>(dst, src, total, K, rs, cs);
}
static void run_wupdate(float* w, const float* nsrc, long cnt, int remap,
                        const float* tao, int taoIdx, float* part, float* red,
                        hipStream_t s) {
  int nb = (int)((cnt + 255) / 256);
  add_reduce<<<nb, 256, 0, s>>>(w, nsrc, cnt, remap, part);
  final_reduce<<<1, 256, 0, s>>>(part, nb, red);
  scale_w<<<nb, 256, 0, s>>>(w, cnt, red, tao, taoIdx, 0.03125f);  // 1024^-0.5
}

extern "C" void kernel_launch(void* const* d_in, const int* in_sizes, int n_in,
                              void* d_out, int out_size, void* d_ws, size_t ws_size,
                              hipStream_t stream) {
  (void)in_sizes; (void)n_in; (void)out_size; (void)ws_size;
  const float* in_x    = (const float*)d_in[0];
  const float* in_cos  = (const float*)d_in[1];
  const float* in_sin  = (const float*)d_in[2];
  const float* in_wq   = (const float*)d_in[3];
  const float* in_ws   = (const float*)d_in[4];
  const float* in_wo   = (const float*)d_in[5];
  const float* in_outw = (const float*)d_in[6];
  const float* in_tao  = (const float*)d_in[7];
  float* out = (float*)d_out;

  char* wsp = (char*)d_ws;
  size_t off = 0;
  auto alloc = [&](size_t bytes) -> void* {
    void* p = wsp + off;
    off = (off + bytes + 255) & ~(size_t)255;
    return p;
  };
  float*  xbuf   = (float*)alloc((size_t)ROWS_ * 1024 * 4);
  bf16_t* xn     = (bf16_t*)alloc((size_t)ROWS_ * 1024 * 2);
  float*  wq     = (float*)alloc((size_t)3072 * 1024 * 4);
  float*  wsw    = (float*)alloc((size_t)2048 * 1024 * 4);
  float*  wo     = (float*)alloc((size_t)1024 * 1024 * 4);
  bf16_t* wq_bf  = (bf16_t*)alloc((size_t)3072 * 1024 * 2);
  bf16_t* ws_bf  = (bf16_t*)alloc((size_t)2048 * 1024 * 2);
  bf16_t* wo_bf  = (bf16_t*)alloc((size_t)1024 * 1024 * 2);
  bf16_t* outw_bf= (bf16_t*)alloc((size_t)1024 * 1024 * 2);
  float*  qkv    = (float*)alloc((size_t)ROWS_ * 3072 * 4);
  float*  wscan  = (float*)alloc((size_t)ROWS_ * 1024 * 4);
  float*  wvscan = (float*)alloc((size_t)ROWS_ * 1024 * 4);
  bf16_t* qh     = (bf16_t*)alloc((size_t)ROWS_ * 1024 * 2);
  bf16_t* kh     = (bf16_t*)alloc((size_t)ROWS_ * 1024 * 2);
  bf16_t* vh     = (bf16_t*)alloc((size_t)ROWS_ * 1024 * 2);
  float*  y      = (float*)alloc((size_t)ROWS_ * 1024 * 4);
  bf16_t* y_bf   = (bf16_t*)alloc((size_t)ROWS_ * 1024 * 2);
  float*  uv     = (float*)alloc((size_t)ROWS_ * 2048 * 4);
  bf16_t* s_bf   = (bf16_t*)alloc((size_t)ROWS_ * 1024 * 2);
  bf16_t* uT_bf  = (bf16_t*)alloc((size_t)1024 * 4096 * 2);
  bf16_t* vT_bf  = (bf16_t*)alloc((size_t)1024 * 4096 * 2);
  float*  ao     = (float*)alloc((size_t)1024 * 1024 * 4);
  bf16_t* aoT_bf = (bf16_t*)alloc((size_t)1024 * 1024 * 2);
  float*  tbig   = (float*)alloc((size_t)3072 * 1024 * 4);
  bf16_t* t_bf   = (bf16_t*)alloc((size_t)3072 * 1024 * 2);
  float*  nbig   = (float*)alloc((size_t)3072 * 1024 * 4);
  bf16_t* wsT_bf = (bf16_t*)alloc((size_t)2048 * 1024 * 2);
  float*  part   = (float*)alloc((size_t)2 * PARTMAX * 4);
  float*  red    = (float*)alloc(256);

  (void)hipMemcpyAsync(xbuf, in_x, (size_t)ROWS_ * 1024 * 4, hipMemcpyDeviceToDevice, stream);
  (void)hipMemcpyAsync(wq, in_wq, (size_t)3072 * 1024 * 4, hipMemcpyDeviceToDevice, stream);
  (void)hipMemcpyAsync(wsw, in_ws, (size_t)2048 * 1024 * 4, hipMemcpyDeviceToDevice, stream);
  (void)hipMemcpyAsync(wo, in_wo, (size_t)1024 * 1024 * 4, hipMemcpyDeviceToDevice, stream);
  run_pack(outw_bf, in_outw, (long)1024 * 1024, 1024, 1024, 1, stream);

  for (int layer = 0; layer < 5; layer++) {
    bool update  = layer < 4;
    bool use_tea = (layer == 3) || (layer == 4);

    run_pack(wq_bf, wq, (long)3072 * 1024, 1024, 1024, 1, stream);
    rmsnorm_rows_bf16<<<ROWS_, 256, 0, stream>>>(xbuf, xn, 1024);
    run_gemm(qkv, xn, wq_bf, ROWS_, 3072, 1024, 1.f, nullptr, stream);

    if (use_tea) {
      tea_prep<<<(ROWS_ * NH_) / 256, 256, 0, stream>>>(qkv, in_cos, in_sin, qh, kh, vh);
      flash_attn<<<dim3(TT_ / 128, NH_, TB_), 256, 0, stream>>>(y, qh, kh, vh);
    } else {
      aft_norm<<<ROWS_, 256, 0, stream>>>(qkv, wscan, wvscan);
      aft_scan<<<(TB_ * 1024) / 256, 256, 0, stream>>>(qkv, wscan, wvscan, y);
    }

    // retention: xo = x + (u*silu(v)) @ wo^T
    run_pack(ws_bf, wsw, (long)2048 * 1024, 1024, 1024, 1, stream);
    run_pack(y_bf, y, (long)ROWS_ * 1024, 1024, 1024, 1, stream);
    run_gemm(uv, y_bf, ws_bf, ROWS_, 2048, 1024, 1.f, nullptr, stream);
    u_silu_pack<<<((long)ROWS_ * 1024) / 256, 256, 0, stream>>>(uv, s_bf);
    run_pack(wo_bf, wo, (long)1024 * 1024, 1024, 1024, 1, stream);
    float* xdst = update ? xbuf : out;
    run_gemm(xdst, s_bf, wo_bf, ROWS_, 1024, 1024, 1.f, xbuf, stream);

    if (update) {
      // ao = rmsnorm(u^T v / (B*sqrt(NE)))
      run_pack(uT_bf, uv, (long)1024 * 4096, 4096, 1, 2048, stream);
      run_pack(vT_bf, uv + 1024, (long)1024 * 4096, 4096, 1, 2048, stream);
      run_gemm(ao, uT_bf, vT_bf, 1024, 1024, 4096, 1.f / 64.f, nullptr, stream);
      rmsnorm_rows_f32<<<1024, 256, 0, stream>>>(ao, 1024);
      run_pack(aoT_bf, ao, (long)1024 * 1024, 1024, 1, 1024, stream);
      // n1 -> wq
      run_gemm(tbig, wq_bf, aoT_bf, 3072, 1024, 1024, 1.f, nullptr, stream);
      silu_pack<<<((long)3072 * 1024) / 256, 256, 0, stream>>>(tbig, t_bf, (long)3072 * 1024);
      run_gemm(nbig, t_bf, outw_bf, 3072, 1024, 1024, 1.f, nullptr, stream);
      run_wupdate(wq, nbig, (long)3072 * 1024, 0, in_tao, 0, part, red, stream);
      // n2 -> ws  (A2 = ws.reshape(NE, 2NQ).T, result remapped via n2.T.reshape)
      run_pack(wsT_bf, wsw, (long)2048 * 1024, 1024, 1, 2048, stream);
      run_gemm(tbig, wsT_bf, aoT_bf, 2048, 1024, 1024, 1.f, nullptr, stream);
      silu_pack<<<((long)2048 * 1024) / 256, 256, 0, stream>>>(tbig, t_bf, (long)2048 * 1024);
      run_gemm(nbig, t_bf, outw_bf, 2048, 1024, 1024, 1.f, nullptr, stream);
      run_wupdate(wsw, nbig, (long)2048 * 1024, 1, in_tao, 1, part, red, stream);
      // n3 -> wo
      run_gemm(tbig, wo_bf, aoT_bf, 1024, 1024, 1024, 1.f, nullptr, stream);
      silu_pack<<<((long)1024 * 1024) / 256, 256, 0, stream>>>(tbig, t_bf, (long)1024 * 1024);
      run_gemm(nbig, t_bf, outw_bf, 1024, 1024, 1024, 1.f, nullptr, stream);
      run_wupdate(wo, nbig, (long)1024 * 1024, 0, in_tao, 2, part, red, stream);
    }
  }
}
